// RecurResNetACT_44023414784271
// MI455X (gfx1250) — matrix operand order repaired and checked by reading, not compiled, not run on hardware
//
#include <hip/hip_runtime.h>
#include <cstdint>

typedef __attribute__((ext_vector_type(16))) __bf16 v16bf;
typedef __attribute__((ext_vector_type(8)))  float  v8f;

__device__ __forceinline__ uint16_t f2bf(float f) {
  uint32_t u = __float_as_uint(f);
  uint32_t r = (u + 0x7FFFu + ((u >> 16) & 1u)) >> 16;
  return (uint16_t)r;
}

union ABFrag { v16bf v; uint32_t u[8]; };

// Async DMA: global -> LDS, 16 bytes, tracked by ASYNCcnt (no VGPR roundtrip).
__device__ __forceinline__ void async_ld_b128(uint32_t lds_off, const void* gaddr) {
  asm volatile("global_load_async_to_lds_b128 %0, %1, off"
               :: "v"(lds_off), "v"(gaddr) : "memory");
}
__device__ __forceinline__ void wait_async0() {
  asm volatile("s_wait_asynccnt 0x0" ::: "memory");
}

// ---------------------------------------------------------------------------
// Weight repack: OIHW f32 [co_real][ci_real][3][3] -> [9][co_pad][ci_pad] bf16
// (zero padded rows/cols so padded WMMA tiles contribute exact zeros)
// ---------------------------------------------------------------------------
__global__ void pack_w_k(const float* __restrict__ w, uint16_t* __restrict__ out,
                         int co_real, int ci_real, int co_pad, int ci_pad) {
  int total = 9 * co_pad * ci_pad;
  for (int idx = blockIdx.x * blockDim.x + threadIdx.x; idx < total;
       idx += gridDim.x * blockDim.x) {
    int ci = idx % ci_pad;
    int t  = idx / ci_pad;
    int co = t % co_pad;
    int r  = t / co_pad;
    float v = 0.f;
    if (co < co_real && ci < ci_real) v = w[(co * ci_real + ci) * 9 + r];
    out[idx] = f2bf(v);
  }
}

// ---------------------------------------------------------------------------
// Implicit-GEMM 3x3 conv (pad=1, stride=1) on NCHW 16x(CI)x32x32 via bf16 WMMA.
// GEMM: M=CO, N=16384 pixels, K=CI*9 (tap-major, ci-minor; K-step = 32).
// 256 threads (8 waves) per 128-pixel strip; wave w owns N-tile w, all CO/16
// M-tiles. Double-buffered LDS; A tile staged with async global->LDS DMA,
// B tile gathered + converted f32->bf16 on the VALU path; f32 accumulate.
// ---------------------------------------------------------------------------
template <int CO, int CI, int CO_STORE, bool RELU, bool RESID>
__global__ __launch_bounds__(256)
void conv3x3_wmma_k(const float* __restrict__ in, const uint16_t* __restrict__ wrep,
                    const float* __restrict__ resid, float* __restrict__ out) {
  constexpr int MT    = CO / 16;         // M tiles per wave
  constexpr int CIB   = (CI + 31) / 32;  // K blocks per tap
  constexpr int CIP   = CIB * 32;        // padded ci per tap in wrep
  constexpr int STEPS = 9 * CIB;
  constexpr int ACH   = CO * 4;          // A tile b128 chunks (CO*32 bf16 / 8)

  __shared__ uint16_t lsA[2][CO * 32];   // [co][k_local], k contiguous
  __shared__ uint16_t lsB[2][128 * 32];  // [p_local][k_local], k contiguous

  const int tid   = threadIdx.x;
  const int wave  = tid >> 5;
  const int lane  = tid & 31;
  const int col   = lane & 15;
  const int khalf = lane >> 4;
  const int p0    = blockIdx.x * 128;
  const int n     = p0 >> 10;            // uniform: one sample per 128-px strip
  const int hw0   = p0 & 1023;

  const uint32_t ldsA0 = (uint32_t)(uintptr_t)&lsA[0][0];
  const uint32_t ldsA1 = (uint32_t)(uintptr_t)&lsA[1][0];

  auto stage = [&](int s, int buf) {
    const int r  = s / CIB, cb = s - r * CIB;
    const int dh = r / 3 - 1, dw = r % 3 - 1;
    // ---- A tile: pure byte-copy of pre-packed bf16 weights -> async DMA
    const uint32_t ldsA = buf ? ldsA1 : ldsA0;
    const uint16_t* wbase = wrep + (size_t)r * CO * CIP + cb * 32;
    for (int c = tid; c < ACH; c += 256) {
      const int co = c >> 2, sub = c & 3;            // 4 chunks of 8 bf16 per row
      async_ld_b128(ldsA + (uint32_t)(c << 4), wbase + (size_t)co * CIP + sub * 8);
    }
    // ---- B tile: shifted tap gather with zero pad, f32 -> bf16
    for (int idx = tid; idx < 128 * 32; idx += 256) {
      const int pl  = idx >> 5;
      const int cil = idx & 31;
      const int ci  = cb * 32 + cil;
      const int hw  = hw0 + pl;
      const int hh  = (hw >> 5) + dh;
      const int ww  = (hw & 31) + dw;
      float v = 0.f;
      if (ci < CI && hh >= 0 && hh < 32 && ww >= 0 && ww < 32)
        v = in[((n * CI + ci) << 10) + (hh << 5) + ww];
      lsB[buf][idx] = f2bf(v);
    }
  };

  v8f acc[MT];
#pragma unroll
  for (int mt = 0; mt < MT; ++mt)
#pragma unroll
    for (int e = 0; e < 8; ++e) acc[mt][e] = 0.f;

  stage(0, 0);
  wait_async0();
  __syncthreads();

  for (int s = 0; s < STEPS; ++s) {
    const int buf = s & 1;
    if (s + 1 < STEPS) stage(s + 1, buf ^ 1);  // overlap DMA/convert with WMMA

    // ---- B fragment (32x16 layout: lanes0-15 K=0..7,16..23, lanes16-31 K=8..15,24..31)
    ABFrag bfr;
    {
      const uint32_t* bp =
          (const uint32_t*)&lsB[buf][((wave << 4) + col) * 32 + (khalf << 3)];
#pragma unroll
      for (int i = 0; i < 8; ++i) bfr.u[i] = bp[i < 4 ? i : i + 4];
    }
    // ---- per M-tile: A fragment (16x32 layout) + WMMA
#pragma unroll
    for (int mt = 0; mt < MT; ++mt) {
      const uint32_t* ap = (const uint32_t*)&lsA[buf][((mt << 4) + col) * 32];
      const int kb = khalf << 2;
      ABFrag afr;
      afr.u[0] = ap[kb + 0];  afr.u[1] = ap[kb + 1];
      afr.u[2] = ap[kb + 2];  afr.u[3] = ap[kb + 3];
      afr.u[4] = ap[kb + 8];  afr.u[5] = ap[kb + 9];
      afr.u[6] = ap[kb + 10]; afr.u[7] = ap[kb + 11];
      acc[mt] = __builtin_amdgcn_wmma_f32_16x16x32_bf16(
          false, afr.v, false, bfr.v, (short)0, acc[mt], false, false);
    }
    wait_async0();   // this wave's next-step A DMA landed in LDS
    __syncthreads(); // all waves done reading buf, next-step tiles visible
  }

  // ---- epilogue: C/D layout VGPR v -> M = v + 8*khalf, N = lane&15
  const int p  = p0 + (wave << 4) + col;
  const int hw = p & 1023;
#pragma unroll
  for (int mt = 0; mt < MT; ++mt) {
#pragma unroll
    for (int v = 0; v < 8; ++v) {
      const int co = (mt << 4) + v + (khalf << 3);
      if (co < CO_STORE) {
        float val = acc[mt][v];
        const int oidx = ((n * CO_STORE + co) << 10) + hw;
        if (RESID) val += resid[oidx];
        if (RELU)  val = fmaxf(val, 0.f);
        out[oidx] = val;
      }
    }
  }
}

// ---------------------------------------------------------------------------
// conv1: 3->128, direct (27 MACs/output, negligible FLOPs) + ReLU
// ---------------------------------------------------------------------------
__global__ void conv1_k(const float* __restrict__ x, const float* __restrict__ w,
                        float* __restrict__ out) {
  int idx = blockIdx.x * blockDim.x + threadIdx.x;
  if (idx >= 16 * 128 * 1024) return;
  const int hw = idx & 1023;
  const int co = (idx >> 10) & 127;
  const int n  = idx >> 17;
  const int h = hw >> 5, ww = hw & 31;
  float s = 0.f;
#pragma unroll
  for (int ci = 0; ci < 3; ++ci)
#pragma unroll
    for (int kh = 0; kh < 3; ++kh)
#pragma unroll
      for (int kw = 0; kw < 3; ++kw) {
        const int hh = h + kh - 1, w2 = ww + kw - 1;
        if (hh >= 0 && hh < 32 && w2 >= 0 && w2 < 32)
          s += w[(co * 3 + ci) * 9 + kh * 3 + kw] *
               x[((n * 3 + ci) << 10) + (hh << 5) + w2];
      }
  out[idx] = fmaxf(s, 0.f);
}

// ---------------------------------------------------------------------------
// Halting unit + ACT state update. One block per batch sample, 128 threads.
// ---------------------------------------------------------------------------
__global__ void halting_k(const float* __restrict__ O,
                          const float* __restrict__ ln_g, const float* __restrict__ ln_b,
                          const float* __restrict__ fc1w, const float* __restrict__ fc1b,
                          const float* __restrict__ fc2w, const float* __restrict__ fc2b,
                          float* __restrict__ acc, float* __restrict__ ponder,
                          float* __restrict__ effw) {
  __shared__ float sv[128];
  __shared__ float sh[64];
  const int n = blockIdx.x, c = threadIdx.x;
  const float* base = O + (((size_t)n * 128 + c) << 10);
  float s = 0.f;
  for (int i = 0; i < 1024; ++i) s += base[i];
  const float v = s * (1.f / 1024.f);
  sv[c] = v;
  __syncthreads();
  float mu = 0.f;
  for (int j = 0; j < 128; ++j) mu += sv[j];
  mu *= (1.f / 128.f);
  float var = 0.f;
  for (int j = 0; j < 128; ++j) { float d = sv[j] - mu; var += d * d; }
  var *= (1.f / 128.f);
  const float vn = (v - mu) * rsqrtf(var + 1e-5f) * ln_g[c] + ln_b[c];
  __syncthreads();
  sv[c] = vn;
  __syncthreads();
  if (c < 64) {
    float t = fc1b[c];
    for (int j = 0; j < 128; ++j) t += sv[j] * fc1w[c * 128 + j];
    sh[c] = fmaxf(t, 0.f);
  }
  __syncthreads();
  if (c == 0) {
    float z = fc2b[0];
    for (int j = 0; j < 64; ++j) z += sh[j] * fc2w[j];
    const float p = 1.f / (1.f + expf(-z));
    const float a = acc[n];
    const float thr = 1.0f - 0.01f;
    const bool still    = a < thr;
    const bool stopping = still && (a + p >= thr);
    const float rem = stopping ? (1.f - a) : p;
    effw[n]    = still ? rem : 0.f;
    ponder[n] += still ? (1.f - p) : 0.f;
    acc[n]     = a + (still ? p : 0.f);
  }
}

__global__ void init_k(float* __restrict__ out, float* __restrict__ acc,
                       float* __restrict__ ponder) {
  int idx = blockIdx.x * blockDim.x + threadIdx.x;
  if (idx < 32769) out[idx] = 0.f;
  if (idx < 16) { acc[idx] = 0.f; ponder[idx] = 0.f; }
}

__global__ void accum_k(const float* __restrict__ cur, const float* __restrict__ effw,
                        float* __restrict__ out) {
  int idx = blockIdx.x * blockDim.x + threadIdx.x;
  if (idx < 32768) {
    const int n = idx >> 11;  // 2*32*32 per sample
    out[idx] += effw[n] * cur[idx];
  }
}

__global__ void final_k(const float* __restrict__ cur, const float* __restrict__ acc,
                        const float* __restrict__ ponder, float* __restrict__ out) {
  int idx = blockIdx.x * blockDim.x + threadIdx.x;
  if (idx < 32768) {
    const int n = idx >> 11;
    const float a = acc[n];
    const float thr = 1.0f - 0.01f;
    const float rem = (a < thr) ? (1.f - a) : 0.f;
    out[idx] += rem * cur[idx];
  }
  if (idx == 0) {
    float s = 0.f;
    for (int j = 0; j < 16; ++j) s += ponder[j];
    out[32768] = s * (1.f / 16.f);
  }
}

// ---------------------------------------------------------------------------
extern "C" void kernel_launch(void* const* d_in, const int* in_sizes, int n_in,
                              void* d_out, int out_size, void* d_ws, size_t ws_size,
                              hipStream_t stream) {
  (void)in_sizes; (void)n_in; (void)out_size; (void)ws_size;
  const float* x    = (const float*)d_in[0];
  const float* w1   = (const float*)d_in[1];
  const float* bw[4] = { (const float*)d_in[2], (const float*)d_in[3],
                         (const float*)d_in[4], (const float*)d_in[5] };
  const float* ln_g = (const float*)d_in[6];
  const float* ln_b = (const float*)d_in[7];
  const float* fc1w = (const float*)d_in[8];
  const float* fc1b = (const float*)d_in[9];
  const float* fc2w = (const float*)d_in[10];
  const float* fc2b = (const float*)d_in[11];
  const float* w2   = (const float*)d_in[12];
  const float* w3   = (const float*)d_in[13];
  const float* w4   = (const float*)d_in[14];

  char* wsp = (char*)d_ws;
  auto wsalloc = [&](size_t bytes) -> char* {
    char* p = wsp;
    wsp += (bytes + 255) & ~((size_t)255);
    return p;
  };
  float* Ob0 = (float*)wsalloc((size_t)16 * 128 * 1024 * 4);
  float* Ob1 = (float*)wsalloc((size_t)16 * 128 * 1024 * 4);
  float* t1  = (float*)wsalloc((size_t)16 * 128 * 1024 * 4);
  float* t2  = (float*)wsalloc((size_t)16 * 32 * 1024 * 4);
  float* t3  = (float*)wsalloc((size_t)16 * 16 * 1024 * 4);
  float* cur = (float*)wsalloc((size_t)16 * 2 * 1024 * 4);
  float* acc = (float*)wsalloc(16 * 4);
  float* pon = (float*)wsalloc(16 * 4);
  float* efw = (float*)wsalloc(16 * 4);
  uint16_t* pwb[4];
  for (int i = 0; i < 4; ++i) pwb[i] = (uint16_t*)wsalloc((size_t)9 * 128 * 128 * 2);
  uint16_t* pw2 = (uint16_t*)wsalloc((size_t)9 * 32 * 128 * 2);
  uint16_t* pw3 = (uint16_t*)wsalloc((size_t)9 * 16 * 32 * 2);
  uint16_t* pw4 = (uint16_t*)wsalloc((size_t)9 * 16 * 32 * 2);

  // ---- repack weights to bf16 [tap][co][ci] (zero padded)
  for (int i = 0; i < 4; ++i)
    pack_w_k<<<(9 * 128 * 128 + 255) / 256, 256, 0, stream>>>(bw[i], pwb[i], 128, 128, 128, 128);
  pack_w_k<<<(9 * 32 * 128 + 255) / 256, 256, 0, stream>>>(w2, pw2, 32, 128, 32, 128);
  pack_w_k<<<(9 * 16 * 32 + 255) / 256, 256, 0, stream>>>(w3, pw3, 8, 32, 16, 32);
  pack_w_k<<<(9 * 16 * 32 + 255) / 256, 256, 0, stream>>>(w4, pw4, 2, 8, 16, 32);

  init_k<<<(32769 + 255) / 256, 256, 0, stream>>>((float*)d_out, acc, pon);

  conv1_k<<<(16 * 128 * 1024 + 255) / 256, 256, 0, stream>>>(x, w1, Ob0);

  float* Ost  = Ob0;
  float* Oalt = Ob1;
  const dim3 cg(128), cb(256);
  const dim3 eg((32768 + 255) / 256), eb(256);

  for (int it = 0; it < 6; ++it) {
    for (int blk = 0; blk < 2; ++blk) {
      conv3x3_wmma_k<128, 128, 128, true, false><<<cg, cb, 0, stream>>>(Ost, pwb[blk * 2 + 0], nullptr, t1);
      conv3x3_wmma_k<128, 128, 128, true, true ><<<cg, cb, 0, stream>>>(t1, pwb[blk * 2 + 1], Ost, Oalt);
      float* tmp = Ost; Ost = Oalt; Oalt = tmp;
    }
    halting_k<<<16, 128, 0, stream>>>(Ost, ln_g, ln_b, fc1w, fc1b, fc2w, fc2b, acc, pon, efw);
    conv3x3_wmma_k<32, 128, 32, true, false><<<cg, cb, 0, stream>>>(Ost, pw2, nullptr, t2);
    conv3x3_wmma_k<16, 32, 16, true, false><<<cg, cb, 0, stream>>>(t2, pw3, nullptr, t3);
    conv3x3_wmma_k<16, 16, 2, false, false><<<cg, cb, 0, stream>>>(t3, pw4, nullptr, cur);
    accum_k<<<eg, eb, 0, stream>>>(cur, efw, (float*)d_out);
  }
  final_k<<<eg, eb, 0, stream>>>(cur, acc, pon, (float*)d_out);
}
